// ManimLLM_27092653703577
// MI455X (gfx1250) — matrix operand order repaired and checked by reading, not compiled, not run on hardware
//
#include <hip/hip_runtime.h>
#include <hip/hip_bf16.h>
#include <math.h>

// Model dims (fixed by the reference)
#define VSZ 32000
#define DM  512
#define NH  8
#define NL  6
#define DFF 2048
#define SEQ 2048
#define DK  64

typedef __bf16 bf16;
typedef __attribute__((ext_vector_type(16))) __bf16 v16bf;
typedef __attribute__((ext_vector_type(8)))  float  v8f;

union FragBf { v16bf v; uint4 q[2]; };

static __device__ inline bf16 f2bf(float f) {
  unsigned u = __builtin_bit_cast(unsigned, f);
  unsigned r = (u + 0x7FFFu + ((u >> 16) & 1u)) >> 16;
  unsigned short s = (unsigned short)r;
  return __builtin_bit_cast(bf16, s);
}

static __device__ inline v8f zero8() {
  v8f z;
#pragma unroll
  for (int i = 0; i < 8; ++i) z[i] = 0.0f;
  return z;
}

// --------------------------------------- transpose-convert: [K][N]f32 -> [N][K]bf16
__global__ void cvt_t_kernel(const float* __restrict__ src, bf16* __restrict__ dst,
                             int K, int N) {
  int i = blockIdx.x * blockDim.x + threadIdx.x;
  int total = K * N;
  int stride = gridDim.x * blockDim.x;
  for (; i < total; i += stride) {
    int k = i / N, n = i - k * N;                 // coalesced read of src
    dst[(size_t)n * K + k] = f2bf(src[i]);
  }
}

// ------------------------------------------------- embedding + positional
__global__ void embed_kernel(const int* __restrict__ ids,
                             const float* __restrict__ emb,
                             float* __restrict__ x) {
  int s = blockIdx.x;
  int tok = ids[s];
  const float sqrtD = 22.627416997969522f;     // sqrt(512)
  const float nl10k = -9.210340371976184f;     // -ln(10000)
  for (int d = threadIdx.x; d < DM; d += blockDim.x) {
    int i2 = (d >> 1) << 1;                    // even index 2i
    float ang = (float)s * __expf(nl10k * (float)i2 / (float)DM);
    float pe = (d & 1) ? __cosf(ang) : __sinf(ang);
    x[(size_t)s * DM + d] = emb[(size_t)tok * DM + d] * sqrtD + pe;
  }
}

// ----------------------------------------------------- bf16 WMMA GEMM
// C[M,N] = A(f32)[M,K] * Bt(bf16, pre-transposed [N][K]) + bias, optional ReLU.
// Block tile 128x128, K-step 32, 256 threads = 8 waves (4 x 2 wave grid),
// each wave owns 32x64 = 2x4 tiles of 16x16 f32 accumulators.
// B tile is fetched with GLOBAL_LOAD_ASYNC_TO_LDS_B128 (ASYNCcnt path);
// A tile is converted f32->bf16 through VALU concurrently with the async copy.
template <int RELU>
__global__ void __launch_bounds__(256)
gemm_bf16_kernel(const float* __restrict__ A, const bf16* __restrict__ Bt,
                 const float* __restrict__ bias, float* __restrict__ C,
                 int M, int N, int K) {
  __shared__ bf16 As[128][40];   // [m][k], pitch 40 halfwords (80B, 16B-aligned rows)
  __shared__ bf16 Bs[128][40];   // [n][k]

  const int tid  = threadIdx.x;
  const int lane = tid & 31;
  const int wave = tid >> 5;
  const int hlf  = (lane >> 4) & 1;
  const int ln16 = lane & 15;
  const int wm   = wave & 3;     // 0..3 : rows
  const int wn   = wave >> 2;    // 0..1 : cols
  const int m0   = blockIdx.y * 128;
  const int n0   = blockIdx.x * 128;

  v8f acc[2][4];
#pragma unroll
  for (int i = 0; i < 2; ++i)
#pragma unroll
    for (int j = 0; j < 4; ++j) acc[i][j] = zero8();

  const int nk = K >> 5;
  for (int kt = 0; kt < nk; ++kt) {
    const int k0 = kt << 5;
    __syncthreads();

    // async-copy B tile: 128 rows x 32 bf16 (64B) = 512 x 16B lane transfers.
    // Bt row (n0+n) bytes [2*(k0+kp) ..) -> Bs[n][kp..kp+8), both 16B-aligned.
#pragma unroll
    for (int it = 0; it < 2; ++it) {
      int i  = tid + it * 256;
      int n  = i >> 2;
      int kp = (i & 3) << 3;                        // 8 bf16 = 16 bytes
      const bf16* g = Bt + (size_t)(n0 + n) * K + k0 + kp;
      unsigned lo = (unsigned)(unsigned long long)(const void*)&Bs[n][kp];
      asm volatile("global_load_async_to_lds_b128 %0, %1, off"
                   :: "v"(lo), "v"(g) : "memory");
    }

    // stage A tile 128x32 (f32 -> bf16) while async B copy is in flight
    {
      int m  = tid >> 1;
      int kc = (tid & 1) << 4;
      const float* src = A + (size_t)(m0 + m) * K + k0 + kc;
#pragma unroll
      for (int j = 0; j < 16; ++j) As[m][kc + j] = f2bf(src[j]);
    }
    if (kt + 1 < nk) {  // prefetch next A K-tile (global_prefetch_b8)
      __builtin_prefetch(A + (size_t)(m0 + (tid >> 1)) * K + k0 + 32 + ((tid & 1) << 4), 0, 1);
    }
    asm volatile("s_wait_asynccnt 0x0" ::: "memory");
    __syncthreads();

    // build fragments per documented 16-bit A/B layouts (wave32)
    FragBf a[2], b[4];
#pragma unroll
    for (int mt = 0; mt < 2; ++mt) {
      int mr = wm * 32 + mt * 16 + ln16;
      a[mt].q[0] = *reinterpret_cast<const uint4*>(&As[mr][hlf * 8]);
      a[mt].q[1] = *reinterpret_cast<const uint4*>(&As[mr][16 + hlf * 8]);
    }
#pragma unroll
    for (int nt = 0; nt < 4; ++nt) {
      int nr = wn * 64 + nt * 16 + ln16;
      b[nt].q[0] = *reinterpret_cast<const uint4*>(&Bs[nr][hlf * 8]);
      b[nt].q[1] = *reinterpret_cast<const uint4*>(&Bs[nr][16 + hlf * 8]);
    }
#pragma unroll
    for (int mt = 0; mt < 2; ++mt)
#pragma unroll
      for (int nt = 0; nt < 4; ++nt)
        acc[mt][nt] = __builtin_amdgcn_wmma_f32_16x16x32_bf16(
            false, a[mt].v, false, b[nt].v, (short)0, acc[mt][nt], false, false);
  }

  // epilogue: bias (+ ReLU) + store f32
#pragma unroll
  for (int mt = 0; mt < 2; ++mt)
#pragma unroll
    for (int nt = 0; nt < 4; ++nt) {
      int col = n0 + wn * 64 + nt * 16 + ln16;
      float bv = bias[col];
#pragma unroll
      for (int r = 0; r < 8; ++r) {
        int row = m0 + wm * 32 + mt * 16 + r + hlf * 8;
        float v = acc[mt][nt][r] + bv;
        if (RELU) v = v > 0.0f ? v : 0.0f;
        C[(size_t)row * N + col] = v;
      }
    }
}

// -------------------------------------------------- flash attention (causal)
// One block = (head h, 64-query tile). 128 threads = 4 waves; wave w owns
// 16 query rows. Online softmax; QK^T and P*V via bf16 WMMA.
__global__ void __launch_bounds__(128)
attn_kernel(const float* __restrict__ Q, const float* __restrict__ Kg,
            const float* __restrict__ Vg, float* __restrict__ O) {
  __shared__ bf16  Qs[64][72];     // [q][dk]
  __shared__ bf16  Ks[64][72];     // [k][dk]
  __shared__ bf16  Vt[64][72];     // [dk][k]  (transposed)
  __shared__ bf16  Ps[64][72];     // [q][k]
  __shared__ float Ss[64][68];     // [q][k] scores f32
  __shared__ float m_s[64], l_s[64], al_s[64];

  const int qb   = blockIdx.x;     // 0..31
  const int h    = blockIdx.y;     // 0..7
  const int q0   = qb * 64;
  const int tid  = threadIdx.x;
  const int lane = tid & 31;
  const int wave = tid >> 5;       // 0..3
  const int hlf  = (lane >> 4) & 1;
  const int ln16 = lane & 15;
  const float scale = 0.125f;      // 1/sqrt(64)

  for (int i = tid; i < 64 * 64; i += 128) {
    int r = i >> 6, c = i & 63;
    Qs[r][c] = f2bf(Q[(size_t)(q0 + r) * DM + h * DK + c]);
  }
  if (tid < 64) { m_s[tid] = -1e30f; l_s[tid] = 0.0f; }

  v8f o[4];
#pragma unroll
  for (int nt = 0; nt < 4; ++nt) o[nt] = zero8();
  __syncthreads();

  for (int kb = 0; kb <= qb; ++kb) {
    const int kbase = kb * 64;
    for (int i = tid; i < 64 * 64; i += 128) {
      int r = i >> 6, c = i & 63;
      float kvK = Kg[(size_t)(kbase + r) * DM + h * DK + c];
      float kvV = Vg[(size_t)(kbase + r) * DM + h * DK + c];
      Ks[r][c] = f2bf(kvK);
      Vt[c][r] = f2bf(kvV);
    }
    __syncthreads();

    // S = Q * K^T (16 rows per wave), DK=64 -> 2 WMMA k-steps
    v8f sc[4];
#pragma unroll
    for (int nt = 0; nt < 4; ++nt) sc[nt] = zero8();
#pragma unroll
    for (int ks = 0; ks < 2; ++ks) {
      const int dk0 = ks * 32;
      FragBf a;
      int mr = wave * 16 + ln16;
      a.q[0] = *reinterpret_cast<const uint4*>(&Qs[mr][dk0 + hlf * 8]);
      a.q[1] = *reinterpret_cast<const uint4*>(&Qs[mr][dk0 + 16 + hlf * 8]);
#pragma unroll
      for (int nt = 0; nt < 4; ++nt) {
        FragBf b;
        int kr = nt * 16 + ln16;
        b.q[0] = *reinterpret_cast<const uint4*>(&Ks[kr][dk0 + hlf * 8]);
        b.q[1] = *reinterpret_cast<const uint4*>(&Ks[kr][dk0 + 16 + hlf * 8]);
        sc[nt] = __builtin_amdgcn_wmma_f32_16x16x32_bf16(
            false, a.v, false, b.v, (short)0, sc[nt], false, false);
      }
    }
#pragma unroll
    for (int nt = 0; nt < 4; ++nt)
#pragma unroll
      for (int r = 0; r < 8; ++r)
        Ss[wave * 16 + r + hlf * 8][nt * 16 + ln16] = sc[nt][r] * scale;
    __syncthreads();

    // online softmax row update (threads 0..63, one row each)
    if (tid < 64) {
      int row = tid;
      int kvalid = (kb == qb) ? (row + 1) : 64;   // causal
      float mx = -1e30f;
      for (int c = 0; c < kvalid; ++c) mx = fmaxf(mx, Ss[row][c]);
      float mold = m_s[row];
      float mnew = fmaxf(mold, mx);
      float alpha = __expf(mold - mnew);
      float sum = 0.0f;
      for (int c = 0; c < 64; ++c) {
        float p = (c < kvalid) ? __expf(Ss[row][c] - mnew) : 0.0f;
        Ps[row][c] = f2bf(p);
        sum += p;
      }
      l_s[row]  = l_s[row] * alpha + sum;
      m_s[row]  = mnew;
      al_s[row] = alpha;
    }
    __syncthreads();

    // rescale O, accumulate O += P * V
#pragma unroll
    for (int nt = 0; nt < 4; ++nt)
#pragma unroll
      for (int r = 0; r < 8; ++r)
        o[nt][r] *= al_s[wave * 16 + r + hlf * 8];
#pragma unroll
    for (int ks = 0; ks < 2; ++ks) {
      FragBf a;
      int mr = wave * 16 + ln16;
      a.q[0] = *reinterpret_cast<const uint4*>(&Ps[mr][ks * 32 + hlf * 8]);
      a.q[1] = *reinterpret_cast<const uint4*>(&Ps[mr][ks * 32 + 16 + hlf * 8]);
#pragma unroll
      for (int nt = 0; nt < 4; ++nt) {
        FragBf b;
        int nc = nt * 16 + ln16;
        b.q[0] = *reinterpret_cast<const uint4*>(&Vt[nc][ks * 32 + hlf * 8]);
        b.q[1] = *reinterpret_cast<const uint4*>(&Vt[nc][ks * 32 + 16 + hlf * 8]);
        o[nt] = __builtin_amdgcn_wmma_f32_16x16x32_bf16(
            false, a.v, false, b.v, (short)0, o[nt], false, false);
      }
    }
    __syncthreads();
  }

  // write ctx = O / l
#pragma unroll
  for (int nt = 0; nt < 4; ++nt)
#pragma unroll
    for (int r = 0; r < 8; ++r) {
      int row = wave * 16 + r + hlf * 8;
      float inv = 1.0f / l_s[row];
      O[(size_t)(q0 + row) * DM + h * DK + nt * 16 + ln16] = o[nt][r] * inv;
    }
}

// --------------------------------------------- residual + LayerNorm (D=512)
template <int RES>
__global__ void __launch_bounds__(256)
ln_kernel(float* __restrict__ x, const float* __restrict__ y,
          const float* __restrict__ g, const float* __restrict__ beta) {
  __shared__ float sh[256];
  const int s = blockIdx.x;
  const int t = threadIdx.x;
  float* xr = x + (size_t)s * DM;
  float a0 = xr[t]       + (RES ? y[(size_t)s * DM + t]       : 0.0f);
  float a1 = xr[t + 256] + (RES ? y[(size_t)s * DM + t + 256] : 0.0f);

  sh[t] = a0 + a1;
  __syncthreads();
  for (int w = 128; w > 0; w >>= 1) { if (t < w) sh[t] += sh[t + w]; __syncthreads(); }
  float mu = sh[0] * (1.0f / 512.0f);
  __syncthreads();

  float d0 = a0 - mu, d1 = a1 - mu;
  sh[t] = d0 * d0 + d1 * d1;
  __syncthreads();
  for (int w = 128; w > 0; w >>= 1) { if (t < w) sh[t] += sh[t + w]; __syncthreads(); }
  float rs = rsqrtf(sh[0] * (1.0f / 512.0f) + 1e-5f);

  xr[t]       = d0 * rs * g[t]       + beta[t];
  xr[t + 256] = d1 * rs * g[t + 256] + beta[t + 256];
}

// ================================================================ launcher
extern "C" void kernel_launch(void* const* d_in, const int* in_sizes, int n_in,
                              void* d_out, int out_size, void* d_ws, size_t ws_size,
                              hipStream_t stream) {
  const int*   ids  = (const int*)  d_in[0];
  const float* emb  = (const float*)d_in[1];
  const float* Wq   = (const float*)d_in[2];
  const float* bq   = (const float*)d_in[3];
  const float* Wk   = (const float*)d_in[4];
  const float* bk   = (const float*)d_in[5];
  const float* Wv   = (const float*)d_in[6];
  const float* bv   = (const float*)d_in[7];
  const float* Wo   = (const float*)d_in[8];
  const float* bo   = (const float*)d_in[9];
  const float* g1   = (const float*)d_in[10];
  const float* be1  = (const float*)d_in[11];
  const float* W1   = (const float*)d_in[12];
  const float* b1   = (const float*)d_in[13];
  const float* W2   = (const float*)d_in[14];
  const float* b2   = (const float*)d_in[15];
  const float* g2   = (const float*)d_in[16];
  const float* be2  = (const float*)d_in[17];
  const float* gf   = (const float*)d_in[18];
  const float* bfp  = (const float*)d_in[19];
  const float* Wout = (const float*)d_in[20];
  const float* bout = (const float*)d_in[21];
  float* out = (float*)d_out;

  // bump allocator on workspace
  char* ws = (char*)d_ws;
  size_t off = 0;
  auto alloc = [&](size_t bytes) -> void* {
    void* p = ws + off;
    off = (off + bytes + 255) & ~(size_t)255;
    return p;
  };

  const size_t szDD = (size_t)NL * DM * DM;
  const size_t szDF = (size_t)NL * DM * DFF;
  bf16* wq_bf   = (bf16*)alloc(szDD * 2);          // all stored transposed [N][K]
  bf16* wk_bf   = (bf16*)alloc(szDD * 2);
  bf16* wv_bf   = (bf16*)alloc(szDD * 2);
  bf16* wo_bf   = (bf16*)alloc(szDD * 2);
  bf16* w1_bf   = (bf16*)alloc(szDF * 2);
  bf16* w2_bf   = (bf16*)alloc(szDF * 2);
  bf16* wout_bf = (bf16*)alloc((size_t)DM * VSZ * 2);
  float* x    = (float*)alloc((size_t)SEQ * DM * 4);
  float* q    = (float*)alloc((size_t)SEQ * DM * 4);
  float* k    = (float*)alloc((size_t)SEQ * DM * 4);
  float* v    = (float*)alloc((size_t)SEQ * DM * 4);
  float* ctx  = (float*)alloc((size_t)SEQ * DM * 4);
  float* ytmp = (float*)alloc((size_t)SEQ * DM * 4);
  float* hbuf = (float*)alloc((size_t)SEQ * DFF * 4);
  (void)ws_size; (void)in_sizes; (void)n_in; (void)out_size;

  // one-time transpose-convert of all weights: [K][N] f32 -> [N][K] bf16
  auto cvtT = [&](const float* s, bf16* d, int K, int N) {
    size_t n = (size_t)K * N;
    int blocks = (int)((n + 255) / 256);
    if (blocks > 4096) blocks = 4096;
    cvt_t_kernel<<<blocks, 256, 0, stream>>>(s, d, K, N);
  };
  for (int l = 0; l < NL; ++l) {
    const size_t oDD = (size_t)l * DM * DM;
    const size_t oDF = (size_t)l * DM * DFF;
    cvtT(Wq + oDD, wq_bf + oDD, DM, DM);
    cvtT(Wk + oDD, wk_bf + oDD, DM, DM);
    cvtT(Wv + oDD, wv_bf + oDD, DM, DM);
    cvtT(Wo + oDD, wo_bf + oDD, DM, DM);
    cvtT(W1 + oDF, w1_bf + oDF, DM, DFF);
    cvtT(W2 + oDF, w2_bf + oDF, DFF, DM);
  }
  cvtT(Wout, wout_bf, DM, VSZ);

  // x = emb[ids]*sqrt(D) + PE
  embed_kernel<<<SEQ, 256, 0, stream>>>(ids, emb, x);

  const dim3 gDD(DM / 128, SEQ / 128);     // (4,16)
  const dim3 gDF(DFF / 128, SEQ / 128);    // (16,16)
  const dim3 gOut(VSZ / 128, SEQ / 128);   // (250,16)
  const dim3 gAttn(SEQ / 64, NH);          // (32,8)

  for (int l = 0; l < NL; ++l) {
    const size_t oDD = (size_t)l * DM * DM;
    const size_t oDF = (size_t)l * DM * DFF;
    gemm_bf16_kernel<0><<<gDD, 256, 0, stream>>>(x, wq_bf + oDD, bq + l * DM, q, SEQ, DM, DM);
    gemm_bf16_kernel<0><<<gDD, 256, 0, stream>>>(x, wk_bf + oDD, bk + l * DM, k, SEQ, DM, DM);
    gemm_bf16_kernel<0><<<gDD, 256, 0, stream>>>(x, wv_bf + oDD, bv + l * DM, v, SEQ, DM, DM);
    attn_kernel<<<gAttn, 128, 0, stream>>>(q, k, v, ctx);
    gemm_bf16_kernel<0><<<gDD, 256, 0, stream>>>(ctx, wo_bf + oDD, bo + l * DM, ytmp, SEQ, DM, DM);
    ln_kernel<1><<<SEQ, 256, 0, stream>>>(x, ytmp, g1 + l * DM, be1 + l * DM);
    gemm_bf16_kernel<1><<<gDF, 256, 0, stream>>>(x, w1_bf + oDF, b1 + l * DFF, hbuf, SEQ, DFF, DM);
    gemm_bf16_kernel<0><<<gDD, 256, 0, stream>>>(hbuf, w2_bf + oDF, b2 + l * DM, ytmp, SEQ, DM, DFF);
    ln_kernel<1><<<SEQ, 256, 0, stream>>>(x, ytmp, g2 + l * DM, be2 + l * DM);
  }
  ln_kernel<0><<<SEQ, 256, 0, stream>>>(x, nullptr, gf, bfp);
  gemm_bf16_kernel<0><<<gOut, 256, 0, stream>>>(x, wout_bf, bout, out, SEQ, VSZ, DM);
}
